// NTXentLoss_10093173146220
// MI455X (gfx1250) — matrix operand order repaired and checked by reading, not compiled, not run on hardware
//
#include <hip/hip_runtime.h>
#include <hip/hip_bf16.h>

typedef __attribute__((ext_vector_type(16))) __bf16 v16bf;
typedef __attribute__((ext_vector_type(8)))  float  v8f;
typedef __attribute__((ext_vector_type(4)))  int    v4i;

#define DIM      256
#define NROWS    8192          // 2B
#define B_SAMP   4096
#define KCHUNKS  8             // 256 / 32
#define EXP2_SCALE 14.4269504088896340736f   // (1/tau) * log2(e)

#define PCOLS    32            // columns per LDS panel buffer
#define PSTRIDE  264           // bf16 per column incl. 8-elem pad (132 dwords == 4 mod 64)
#define PANEL_BYTES (PCOLS * PSTRIDE * 2)    // 16896
#define GPANEL_BYTES (PCOLS * DIM * 2)       // 16384 global bytes per panel
#define CQUARTS  4
#define QCOLS    (NROWS / CQUARTS)           // 2048 columns per quarter
#define PANELS   (QCOLS / PCOLS)             // 64 panels per block
#define CHUNKS_PER_THREAD 4

#if __has_builtin(__builtin_amdgcn_global_load_async_to_lds_b128)
#define USE_ASYNC_LDS 1
#else
#define USE_ASYNC_LDS 0
#endif

union Frag16 { uint4 q[2]; v16bf v; };

__device__ __forceinline__ unsigned short f2bf_rne(float f) {
    unsigned u = __float_as_uint(f);
    unsigned r = u + 0x7FFFu + ((u >> 16) & 1u);
    return (unsigned short)(r >> 16);
}

__device__ __forceinline__ float dot4(float4 a, float4 b) {
    return a.x * b.x + a.y * b.y + a.z * b.z + a.w * b.w;
}

template <int N>
__device__ __forceinline__ void wait_asynccnt() {
#if __has_builtin(__builtin_amdgcn_s_wait_asynccnt)
    __builtin_amdgcn_s_wait_asynccnt(N);
#else
    asm volatile("s_wait_asynccnt %0" :: "i"(N) : "memory");
#endif
}

// ---------------------------------------------------------------------------
// Kernel 1: per-row L2 normalize, write bf16 rows + inverse norms.
// ---------------------------------------------------------------------------
__global__ void ntx_normalize(const float* __restrict__ z1,
                              const float* __restrict__ z2,
                              unsigned short* __restrict__ zbf,
                              float* __restrict__ invn) {
    const int wave = threadIdx.x >> 5;
    const int lane = threadIdx.x & 31;
    const int row  = blockIdx.x * 8 + wave;           // 0..8191

    const float* src = (row < B_SAMP) ? (z1 + (size_t)row * DIM)
                                      : (z2 + (size_t)(row - B_SAMP) * DIM);
    const float4* s4 = (const float4*)src;
    float4 v0 = s4[lane];
    float4 v1 = s4[lane + 32];

    float ss = dot4(v0, v0) + dot4(v1, v1);
    #pragma unroll
    for (int m = 16; m >= 1; m >>= 1) ss += __shfl_xor(ss, m, 32);

    const float inv = 1.0f / fmaxf(sqrtf(ss), 1e-12f);
    if (lane == 0) invn[row] = inv;

    uint2* dst = (uint2*)zbf;
    uint2 p0, p1;
    p0.x = (unsigned)f2bf_rne(v0.x * inv) | ((unsigned)f2bf_rne(v0.y * inv) << 16);
    p0.y = (unsigned)f2bf_rne(v0.z * inv) | ((unsigned)f2bf_rne(v0.w * inv) << 16);
    p1.x = (unsigned)f2bf_rne(v1.x * inv) | ((unsigned)f2bf_rne(v1.y * inv) << 16);
    p1.y = (unsigned)f2bf_rne(v1.z * inv) | ((unsigned)f2bf_rne(v1.w * inv) << 16);
    dst[(size_t)row * 64 + lane]      = p0;
    dst[(size_t)row * 64 + 32 + lane] = p1;
}

// ---------------------------------------------------------------------------
// Kernel 2: exact fp32 positive logit s_i = <z1_i,z2_i> * inv1 * inv2 / tau
// ---------------------------------------------------------------------------
__global__ void ntx_sim(const float* __restrict__ z1,
                        const float* __restrict__ z2,
                        const float* __restrict__ invn,
                        float* __restrict__ sim) {
    const int wave = threadIdx.x >> 5;
    const int lane = threadIdx.x & 31;
    const int i    = blockIdx.x * 8 + wave;           // 0..4095

    const float4* a = (const float4*)(z1 + (size_t)i * DIM);
    const float4* b = (const float4*)(z2 + (size_t)i * DIM);
    float d = dot4(a[lane], b[lane]) + dot4(a[lane + 32], b[lane + 32]);
    #pragma unroll
    for (int m = 16; m >= 1; m >>= 1) d += __shfl_xor(d, m, 32);

    if (lane == 0)
        sim[i] = d * invn[i] * invn[i + B_SAMP] * 10.0f;   // 1/tau
}

// ---------------------------------------------------------------------------
// Kernel 3: partial[q][r] = sum_{c in quarter q} exp(<zhat_r, zhat_c>/tau).
// 8 waves x private 16-row strips (128 rows/block); 32-col bf16 panel double-
// buffered in LDS, filled by global_load_async_to_lds (ASYNCcnt).
// All hot-loop addresses = one base VGPR + compile-time instruction offsets.
// Grid = 64 row-groups x 4 column-quarters = 256 blocks. No atomics.
// ---------------------------------------------------------------------------
__global__ void ntx_rowsum(const unsigned short* __restrict__ zbf,
                           float* __restrict__ partial) {
    const int tid  = threadIdx.x;
    const int wave = tid >> 5;
    const int lane = tid & 31;
    const int half = lane >> 4;          // 0: lanes 0-15, 1: lanes 16-31
    const int m16  = lane & 15;

    const int rg    = blockIdx.x >> 2;   // row group: 128 rows
    const int cq    = blockIdx.x & 3;    // column quarter
    const int rbase = rg * 128 + wave * 16;
    const int C0    = cq * QCOLS;

    const uint4* zq = (const uint4*)zbf;     // 8 bf16 per uint4

    __shared__ unsigned short panel[2][PCOLS * PSTRIDE];   // 2 x 16.5 KB

    // ---- A fragments: this wave's 16 rows, all 8 K-chunks, in registers.
    Frag16 a[KCHUNKS];
    {
        const size_t arow = (size_t)(rbase + m16) * DIM;
        const int h = half * 16;
        #pragma unroll
        for (int kc = 0; kc < KCHUNKS; ++kc) {
            const int k0 = kc * 32 + h;
            a[kc].q[0] = zq[(arow + k0) >> 3];
            a[kc].q[1] = zq[(arow + k0 + 8) >> 3];
        }
    }

    float rsum[8];
    #pragma unroll
    for (int m = 0; m < 8; ++m) rsum[m] = 0.0f;

    const int kb = half * 16;            // B: lanes<16 -> K 0-15, lanes>=16 -> K 16-31

    // ---- staging layout: thread t owns 64 contiguous bytes of one column.
    //      col = t>>3, byte offset = (t&7)*64 + 16*i  (same offset on both sides)
    const int scol = tid >> 3;
    const int soff = (tid & 7) * 64;
    const char* gstage  = (const char*)zbf + (size_t)(C0 + scol) * (DIM * 2) + soff;
    const int   lstoff  = scol * (PSTRIDE * 2) + soff;
    char* const lstage0 = (char*)&panel[0][0] + lstoff;

    // ---- B-read base: one per-lane pointer; all step offsets are constants.
    const char* const bb0 = (const char*)&panel[0][0] + (m16 * PSTRIDE + kb) * 2;

#if USE_ASYNC_LDS
    #define ISSUE_PANEL(P, BUF)                                                    \
        {                                                                          \
            const char* g_ = gstage + (size_t)(P) * GPANEL_BYTES;                  \
            char* l_ = lstage0 + (BUF) * PANEL_BYTES;                              \
            __builtin_amdgcn_global_load_async_to_lds_b128(                        \
                (__attribute__((address_space(1))) v4i*)g_,                        \
                (__attribute__((address_space(3))) v4i*)l_, 0, 0);                 \
            __builtin_amdgcn_global_load_async_to_lds_b128(                        \
                (__attribute__((address_space(1))) v4i*)g_,                        \
                (__attribute__((address_space(3))) v4i*)l_, 16, 0);                \
            __builtin_amdgcn_global_load_async_to_lds_b128(                        \
                (__attribute__((address_space(1))) v4i*)g_,                        \
                (__attribute__((address_space(3))) v4i*)l_, 32, 0);                \
            __builtin_amdgcn_global_load_async_to_lds_b128(                        \
                (__attribute__((address_space(1))) v4i*)g_,                        \
                (__attribute__((address_space(3))) v4i*)l_, 48, 0);                \
        }
    ISSUE_PANEL(0, 0)
#else
    uint4 stage[CHUNKS_PER_THREAD];
    #define LOAD_STAGE(P)                                                          \
        {                                                                          \
            const char* g_ = gstage + (size_t)(P) * GPANEL_BYTES;                  \
            _Pragma("unroll")                                                      \
            for (int i_ = 0; i_ < CHUNKS_PER_THREAD; ++i_)                         \
                stage[i_] = *(const uint4*)(g_ + 16 * i_);                         \
        }
    #define STORE_STAGE(BUF)                                                       \
        {                                                                          \
            char* l_ = lstage0 + (BUF) * PANEL_BYTES;                              \
            _Pragma("unroll")                                                      \
            for (int i_ = 0; i_ < CHUNKS_PER_THREAD; ++i_)                         \
                *(uint4*)(l_ + 16 * i_) = stage[i_];                               \
        }
    LOAD_STAGE(0)
    STORE_STAGE(0)
    __syncthreads();
#endif

    for (int p = 0; p < PANELS; ++p) {
        const int cur = p & 1;
#if USE_ASYNC_LDS
        if (p + 1 < PANELS) {
            ISSUE_PANEL(p + 1, cur ^ 1)           // buffer last read in p-1 (safe)
            wait_asynccnt<CHUNKS_PER_THREAD>();   // my panel-p chunks have landed
        } else {
            wait_asynccnt<0>();
        }
        __syncthreads();                          // everyone's panel-p chunks visible
#else
        if (p + 1 < PANELS) LOAD_STAGE(p + 1)
#endif

        // ---- compute: 2 column tiles, interleaved WMMA chains, 3-deep B pipeline.
        // address = bbase + constant -> folds into ds_load_b128 offset field.
        const char* bbase = bb0 + cur * PANEL_BYTES;
        v8f acc0 = {}, acc1 = {};
        #define LOAD_B(DST, S)                                                     \
            {                                                                      \
                const int off_ = ((S) & 1) * (16 * PSTRIDE * 2) + ((S) >> 1) * 64; \
                DST.q[0] = *(const uint4*)(bbase + off_);                          \
                DST.q[1] = *(const uint4*)(bbase + off_ + 16);                     \
            }
        Frag16 b0, b1, b2;
        LOAD_B(b0, 0)
        LOAD_B(b1, 1)
        #pragma unroll
        for (int s = 0; s < 16; ++s) {       // step s: tile = s&1, kchunk = s>>1
            if (s + 2 < 16) { LOAD_B(b2, s + 2) }
            const int kc = s >> 1;
            if ((s & 1) == 0)
                acc0 = __builtin_amdgcn_wmma_f32_16x16x32_bf16(
                        false, a[kc].v, false, b0.v, (short)0, acc0, false, false);
            else
                acc1 = __builtin_amdgcn_wmma_f32_16x16x32_bf16(
                        false, a[kc].v, false, b0.v, (short)0, acc1, false, false);
            b0 = b1; b1 = b2;
        }
        #pragma unroll
        for (int m = 0; m < 8; ++m)
            rsum[m] += exp2f(acc0[m] * EXP2_SCALE) + exp2f(acc1[m] * EXP2_SCALE);

#if USE_ASYNC_LDS
        __syncthreads();                     // compute(p) done before p+1 overwrites buf[cur]
#else
        if (p + 1 < PANELS) STORE_STAGE(cur ^ 1)
        __syncthreads();
#endif
    }

    // reduce across the 16-lane group; each wave owns its rows -> direct store
    #pragma unroll
    for (int m = 0; m < 8; ++m) {
        float v = rsum[m];
        v += __shfl_xor(v, 1, 32);
        v += __shfl_xor(v, 2, 32);
        v += __shfl_xor(v, 4, 32);
        v += __shfl_xor(v, 8, 32);
        rsum[m] = v;
    }
    if (m16 == 0) {
        float* dst = partial + (size_t)cq * NROWS + rbase + half * 8;
        #pragma unroll
        for (int m = 0; m < 8; ++m) dst[m] = rsum[m];
    }
}

// ---------------------------------------------------------------------------
// Kernel 4: combine quarters; out[i] = (log(S_i-p) + log(S_{i+B}-p) - 2 s)/2B
// ---------------------------------------------------------------------------
__global__ void ntx_finish(const float* __restrict__ partial,
                           const float* __restrict__ sim,
                           float* __restrict__ out) {
    const int i = blockIdx.x * blockDim.x + threadIdx.x;
    if (i < B_SAMP) {
        float sa = 0.0f, sb = 0.0f;
        #pragma unroll
        for (int q = 0; q < CQUARTS; ++q) {
            sa += partial[(size_t)q * NROWS + i];
            sb += partial[(size_t)q * NROWS + i + B_SAMP];
        }
        const float s = sim[i];
        const float p = __expf(s);
        out[i] = (__logf(sa - p) + __logf(sb - p) - 2.0f * s) * (1.0f / (float)NROWS);
    }
}

extern "C" void kernel_launch(void* const* d_in, const int* in_sizes, int n_in,
                              void* d_out, int out_size, void* d_ws, size_t ws_size,
                              hipStream_t stream) {
    const float* z1 = (const float*)d_in[0];
    const float* z2 = (const float*)d_in[1];
    float* out = (float*)d_out;

    char* ws = (char*)d_ws;
    unsigned short* zbf = (unsigned short*)ws;                   // 4 MiB bf16 Z
    float* invn    = (float*)(ws + (size_t)NROWS * DIM * 2);     // 8192 f32
    float* partial = invn + NROWS;                               // 4*8192 f32
    float* sim     = partial + (size_t)CQUARTS * NROWS;          // 4096 f32

    ntx_normalize<<<NROWS / 8, 256, 0, stream>>>(z1, z2, zbf, invn);
    ntx_sim<<<B_SAMP / 8, 256, 0, stream>>>(z1, z2, invn, sim);
    ntx_rowsum<<<64 * CQUARTS, 256, 0, stream>>>(zbf, partial);
    ntx_finish<<<(B_SAMP + 255) / 256, 256, 0, stream>>>(partial, sim, out);
}
